// NuModelII_53506702573825
// MI455X (gfx1250) — matrix operand order repaired and checked by reading, not compiled, not run on hardware
//
#include <hip/hip_runtime.h>

// ---------------------------------------------------------------------------
// NuModelII: 5 chained Mamba selectors on MI455X (gfx1250).
//  - GEMMs: f16 WMMA (v_wmma_f32_16x16x32_f16), register-blocked 32x64/wave
//    (spill-free), compile-time N/K so all fragment loads / C stores are a
//    single base pointer + immediate offset.
//  - Scan: async global->LDS broadcast of the per-timestep (dt,B,C) row.
//  - Conv: rolling-window depthwise conv, each input read once.
// Workspace need: ~117 MB.
// ---------------------------------------------------------------------------

typedef __attribute__((ext_vector_type(16))) _Float16 v16h;
typedef __attribute__((ext_vector_type(8)))  _Float16 v8h;
typedef __attribute__((ext_vector_type(8)))  float    v8f;

constexpr int BATCH = 16;
constexpr int SEQ   = 512;
constexpr int DM    = 256;   // d_model
constexpr int DI    = 1024;  // d_inner
constexpr int DSTATE= 16;
constexpr int DTR   = 16;
constexpr int ROWS  = BATCH * SEQ;          // 8192 token rows
constexpr int XDBCW = DTR + 2 * DSTATE;     // 48

// ---- workspace layout (bytes) ----
constexpr size_t OFF_H16  = 0;                                   // ROWS*DM f16   (4 MB)
constexpr size_t OFF_XZ   = OFF_H16  + (size_t)ROWS*DM*2;        // ROWS*2*DI f32 (64 MB)
constexpr size_t OFF_U16  = OFF_XZ   + (size_t)ROWS*2*DI*4;      // ROWS*DI f16   (16 MB)
constexpr size_t OFF_XDBC = OFF_U16  + (size_t)ROWS*DI*2;        // ROWS*48 f32   (1.5 MB)
constexpr size_t OFF_Y16  = OFF_XDBC + (size_t)ROWS*XDBCW*4;     // ROWS*DI f16   (16 MB)
constexpr size_t OFF_YO   = OFF_Y16  + (size_t)ROWS*DI*2;        // ROWS*DM f32   (8 MB)
constexpr size_t OFF_WIN  = OFF_YO   + (size_t)ROWS*DM*4;        // 2*DI*DM f16   (1 MB)
constexpr size_t OFF_WXP  = OFF_WIN  + (size_t)2*DI*DM*2;        // 48*DI f16
constexpr size_t OFF_WOUT = OFF_WXP  + (size_t)XDBCW*DI*2;       // DM*DI f16

static __device__ __forceinline__ float sigmoidf_(float x) {
  return 1.0f / (1.0f + __expf(-x));
}

// ---------------------------------------------------------------------------
// f32 -> f16 weight conversion
// ---------------------------------------------------------------------------
__global__ void k_cvt_f16(const float* __restrict__ in, _Float16* __restrict__ out, int n) {
  int i = blockIdx.x * blockDim.x + threadIdx.x;
  if (i < n) out[i] = (_Float16)in[i];
}

// ---------------------------------------------------------------------------
// Embed: h = sigmoid(x_cat @ W1^T + b1), x_cat = [x(B,L,3), o1.., o_{s-1}]
// (prior outputs read directly from d_out: feature f>=3 == dout[b*64 + f-3])
// ---------------------------------------------------------------------------
__global__ void k_embed(const float* __restrict__ x,    // (B,L,3)
                        const float* __restrict__ W1,   // (DM, F)
                        const float* __restrict__ b1,   // (DM)
                        const float* __restrict__ dout, // (B,64)
                        _Float16* __restrict__ h16,     // (ROWS, DM)
                        int F) {
  int idx = blockIdx.x * blockDim.x + threadIdx.x;
  if (idx >= ROWS * DM) return;
  int d   = idx & (DM - 1);
  int row = idx >> 8;          // /DM
  int b   = row / SEQ;
  const float* w  = W1 + (size_t)d * F;
  const float* xp = x + (size_t)row * 3;
  float acc = b1[d] + w[0]*xp[0] + w[1]*xp[1] + w[2]*xp[2];
  for (int f = 3; f < F; ++f) acc += w[f] * dout[b * 64 + (f - 3)];
  h16[idx] = (_Float16)sigmoidf_(acc);
}

// ---------------------------------------------------------------------------
// WMMA NT-GEMM: C(MxNc) = A(MxKc) * B(NcxKc)^T, A/B f16 K-contiguous, C f32.
// Register-blocked MT x NT 16x16 tiles per wave; Nc/Kc compile-time so every
// fragment load / C store is base pointer + immediate offset. <2,4> = 32x64
// per wave: 12 b128 loads per 8 WMMA, ~140 VGPRs (no spills).
// Fragment layouts per CDNA5 ISA 7.12.2.
// ---------------------------------------------------------------------------
template <int MT, int NT, int Nc, int Kc>
__global__ void k_gemm_nt(const _Float16* __restrict__ A,
                          const _Float16* __restrict__ Bm,
                          float* __restrict__ C) {
  const int lane   = threadIdx.x & 31;
  const int wave   = threadIdx.x >> 5;
  const int nwaves = blockDim.x >> 5;
  const int ntile0 = (blockIdx.y * nwaves + wave) * NT;
  if ((ntile0 + NT) * 16 > Nc) return;      // wave-uniform guard (EXEC all-1s)
  const int m0 = blockIdx.x * (16 * MT);
  const int lg = lane >> 4;                 // lane-group 0/1
  const int lr = lane & 15;

  // Single base pointer per operand; mi/ni/k offsets are immediates.
  const _Float16* ap = A  + (size_t)(m0 + lr) * Kc + lg * 8;        // K = lg*8 + {0..7,16..23}
  const _Float16* bp = Bm + (size_t)(ntile0 * 16 + lr) * Kc + lg * 8;  // K = lg*8 + {0..7,16..23}

  __builtin_prefetch(ap, 0, 0);             // global_prefetch_b8 (L2 warm)
  __builtin_prefetch(bp, 0, 0);

  v8f acc[MT][NT] = {};
  for (int k0 = 0; k0 < Kc; k0 += 32) {
    v16h a[MT];
#pragma unroll
    for (int mi = 0; mi < MT; ++mi) {
      v8h alo = *(const v8h*)(ap + mi * 16 * Kc + k0);
      v8h ahi = *(const v8h*)(ap + mi * 16 * Kc + k0 + 16);
#pragma unroll
      for (int i = 0; i < 8; ++i) { a[mi][i] = alo[i]; a[mi][i + 8] = ahi[i]; }
    }
#pragma unroll
    for (int ni = 0; ni < NT; ++ni) {
      v8h blo = *(const v8h*)(bp + ni * 16 * Kc + k0);
      v8h bhi = *(const v8h*)(bp + ni * 16 * Kc + k0 + 16);
      v16h b;
#pragma unroll
      for (int i = 0; i < 8; ++i) { b[i] = blo[i]; b[i + 8] = bhi[i]; }
#pragma unroll
      for (int mi = 0; mi < MT; ++mi)
        acc[mi][ni] = __builtin_amdgcn_wmma_f32_16x16x32_f16(
            false, a[mi], false, b, (short)0, acc[mi][ni], false, false);
    }
  }
  // C layout: VGPR r, lane l -> (M = r + 8*lg, N = lr)
  float* cp = C + (size_t)(m0 + lg * 8) * Nc + ntile0 * 16 + lr;
#pragma unroll
  for (int mi = 0; mi < MT; ++mi)
#pragma unroll
    for (int ni = 0; ni < NT; ++ni)
#pragma unroll
      for (int r = 0; r < 8; ++r)
        cp[(size_t)(mi * 16 + r) * Nc + ni * 16] = acc[mi][ni][r];
}

// ---------------------------------------------------------------------------
// Causal depthwise conv1d (D_CONV=8) + bias + SiLU on xm = xz[:, :DI].
// One thread per (b, d) channel, rolling 8-deep register window over time:
// every xm element is read from memory exactly once.
// ---------------------------------------------------------------------------
__global__ void k_conv_silu(const float* __restrict__ xz,  // (ROWS, 2*DI)
                            const float* __restrict__ cw,  // (DI, 8)
                            const float* __restrict__ cb,  // (DI)
                            _Float16* __restrict__ u16) {  // (ROWS, DI)
  const int b = blockIdx.y;
  const int d = blockIdx.x * blockDim.x + threadIdx.x;     // < DI
  float w[8], win[8];
#pragma unroll
  for (int t = 0; t < 8; ++t) { w[t] = cw[(size_t)d * 8 + t]; win[t] = 0.0f; }
  const float bias = cb[d];
  for (int l = 0; l < SEQ; ++l) {
#pragma unroll
    for (int t = 0; t < 7; ++t) win[t] = win[t + 1];
    win[7] = xz[(size_t)(b * SEQ + l) * (2 * DI) + d];
    float acc = bias;
#pragma unroll
    for (int t = 0; t < 8; ++t) acc += w[t] * win[t];
    u16[(size_t)(b * SEQ + l) * DI + d] = (_Float16)(acc * sigmoidf_(acc));
  }
}

// ---------------------------------------------------------------------------
// Selective scan: one thread per (b, d) channel, 16 states in VGPRs.
// The per-timestep 48-float (dt,B,C) row is broadcast to the block through
// LDS using the CDNA5 async copy path: lanes 0..11 issue
// GLOBAL_LOAD_ASYNC_TO_LDS_B128 (ASYNCcnt), issuing wave waits asynccnt 0,
// block barrier publishes the row. Produces gated y (f16) for out_proj.
// ---------------------------------------------------------------------------
__global__ void k_scan(const float* __restrict__ xz,     // (ROWS, 2*DI) - z half
                       const _Float16* __restrict__ u16, // (ROWS, DI)
                       const float* __restrict__ xdbc,   // (ROWS, 48)
                       const float* __restrict__ dtw,    // (DI, DTR)
                       const float* __restrict__ dtb,    // (DI)
                       const float* __restrict__ Alog,   // (DI, 16)
                       const float* __restrict__ Dp,     // (DI)
                       _Float16* __restrict__ y16) {     // (ROWS, DI)
  __shared__ float srow[XDBCW];
  const int b = blockIdx.y;
  const int d = blockIdx.x * blockDim.x + threadIdx.x;   // < DI
  // LDS byte offset of srow for this thread's async-copy slot (generic
  // LDS-aperture pointers carry the LDS offset in their low 32 bits).
  const unsigned ldsOff = (unsigned)(uintptr_t)(&srow[0]) + (threadIdx.x << 4);

  float An[DSTATE], Wt[DTR], hs[DSTATE];
#pragma unroll
  for (int n = 0; n < DSTATE; ++n) {
    An[n] = -__expf(Alog[(size_t)d * DSTATE + n]);
    Wt[n] = dtw[(size_t)d * DTR + n];
    hs[n] = 0.0f;
  }
  const float db = dtb[d];
  const float Dd = Dp[d];

  for (int l = 0; l < SEQ; ++l) {
    const size_t row = (size_t)b * SEQ + l;
    __syncthreads();                          // previous step's reads done
    if (threadIdx.x < (XDBCW / 4)) {          // 12 lanes x 16B = 192B row
      const float* gp = xdbc + row * XDBCW + (threadIdx.x << 2);
      asm volatile("global_load_async_to_lds_b128 %0, %1, off"
                   :: "v"(ldsOff), "v"(gp) : "memory");
    }
    asm volatile("s_wait_asynccnt 0x0" ::: "memory"); // no-op in non-issuing waves
    __syncthreads();

    float dtacc = db;
#pragma unroll
    for (int r = 0; r < DTR; ++r) dtacc += srow[r] * Wt[r];
    float delta = (dtacc > 20.0f) ? dtacc : log1pf(__expf(dtacc));   // softplus

    float u  = (float)u16[row * DI + d];
    float du = delta * u;
    float y  = 0.0f;
#pragma unroll
    for (int n = 0; n < DSTATE; ++n) {
      float dA = __expf(delta * An[n]);
      hs[n] = dA * hs[n] + du * srow[DTR + n];          // B row
      y += hs[n] * srow[DTR + DSTATE + n];              // C row
    }
    float z = xz[row * (2 * DI) + DI + d];
    float out = (y + u * Dd) * (z * sigmoidf_(z));      // + u*D, gate silu(z)
    y16[row * DI + d] = (_Float16)out;
  }
}

// ---------------------------------------------------------------------------
// Head: logits of the LAST timestep only, /T, softmax -> d_out columns.
// One wave (32 lanes) per batch row; wave32 shuffle reductions.
// ---------------------------------------------------------------------------
__global__ void k_head(const float* __restrict__ yo,   // (ROWS, DM)
                       const float* __restrict__ W2,   // (outs, DM)
                       const float* __restrict__ b2,   // (outs)
                       const float* __restrict__ temp, // scalar
                       float* __restrict__ dout,       // (B, 64)
                       int outs, int colOff) {
  const int b = blockIdx.x;
  const int o = threadIdx.x;                  // 0..31
  float lg = -3.402823466e38f;
  if (o < outs) {
    float acc = b2[o];
    const float* yp = yo + ((size_t)b * SEQ + (SEQ - 1)) * DM;
    const float* wp = W2 + (size_t)o * DM;
    for (int k = 0; k < DM; ++k) acc += yp[k] * wp[k];
    lg = acc / temp[0];
  }
  float mx = lg;
#pragma unroll
  for (int s = 16; s > 0; s >>= 1) mx = fmaxf(mx, __shfl_xor(mx, s, 32));
  float e = (o < outs) ? __expf(lg - mx) : 0.0f;
  float sum = e;
#pragma unroll
  for (int s = 16; s > 0; s >>= 1) sum += __shfl_xor(sum, s, 32);
  if (o < outs) dout[b * 64 + colOff + o] = e / sum;
}

// ---------------------------------------------------------------------------
// Launch: 5 selectors, stream-ordered (each reads prior outputs from d_out).
// ---------------------------------------------------------------------------
extern "C" void kernel_launch(void* const* d_in, const int* in_sizes, int n_in,
                              void* d_out, int out_size, void* d_ws, size_t ws_size,
                              hipStream_t stream) {
  (void)in_sizes; (void)n_in; (void)out_size; (void)ws_size;
  const float* x  = (const float*)d_in[0];
  const float* temp = (const float*)d_in[1 + 5 * 13];   // after 5 param dicts
  float* dout = (float*)d_out;

  char* ws = (char*)d_ws;
  _Float16* h16   = (_Float16*)(ws + OFF_H16);
  float*    xz    = (float*)   (ws + OFF_XZ);
  _Float16* u16   = (_Float16*)(ws + OFF_U16);
  float*    xdbc  = (float*)   (ws + OFF_XDBC);
  _Float16* y16   = (_Float16*)(ws + OFF_Y16);
  float*    yo    = (float*)   (ws + OFF_YO);
  _Float16* win16 = (_Float16*)(ws + OFF_WIN);
  _Float16* wxp16 = (_Float16*)(ws + OFF_WXP);
  _Float16* wout16= (_Float16*)(ws + OFF_WOUT);

  const int OUTSV[5] = {25, 3, 4, 16, 16};
  const int FS[5]    = {3, 28, 31, 35, 51};   // 3 + cumulative prior outs
  const int COFF[5]  = {0, 25, 28, 32, 48};   // column offset in d_out row

  for (int s = 0; s < 5; ++s) {
    const int base = 1 + s * 13;   // params: W1,b1,in_proj,conv_w,conv_b,x_proj,
                                   //         dt_w,dt_b,A_log,D,out_proj,W2,b2
    const float* W1     = (const float*)d_in[base + 0];
    const float* b1     = (const float*)d_in[base + 1];
    const float* inproj = (const float*)d_in[base + 2];
    const float* convw  = (const float*)d_in[base + 3];
    const float* convb  = (const float*)d_in[base + 4];
    const float* xproj  = (const float*)d_in[base + 5];
    const float* dt_w   = (const float*)d_in[base + 6];
    const float* dt_b   = (const float*)d_in[base + 7];
    const float* A_log  = (const float*)d_in[base + 8];
    const float* Dpar   = (const float*)d_in[base + 9];
    const float* outproj= (const float*)d_in[base + 10];
    const float* W2     = (const float*)d_in[base + 11];
    const float* b2     = (const float*)d_in[base + 12];

    // weight f16 conversion (reused buffers, re-done each selector)
    k_cvt_f16<<<(2 * DI * DM + 255) / 256, 256, 0, stream>>>(inproj,  win16,  2 * DI * DM);
    k_cvt_f16<<<(XDBCW * DI + 255) / 256, 256, 0, stream>>>(xproj,   wxp16,  XDBCW * DI);
    k_cvt_f16<<<(DM * DI + 255) / 256, 256, 0, stream>>>(outproj, wout16, DM * DI);

    // embed + sigmoid -> f16 activations
    k_embed<<<(ROWS * DM + 255) / 256, 256, 0, stream>>>(x, W1, b1, dout, h16, FS[s]);

    // xz = h @ in_proj^T   (8192 x 2048 x 256): 32x64 per wave, 4 waves/block
    k_gemm_nt<2, 4, 2 * DI, DM><<<dim3(ROWS / 32, (2 * DI) / 256), 128, 0, stream>>>(
        h16, win16, xz);

    // causal depthwise conv + SiLU -> u (f16)
    k_conv_silu<<<dim3(DI / 256, BATCH), 256, 0, stream>>>(xz, convw, convb, u16);

    // xdbc = u @ x_proj^T  (8192 x 48 x 1024): one wave covers all 48 cols
    k_gemm_nt<2, 3, XDBCW, DI><<<dim3(ROWS / 32, 1), 32, 0, stream>>>(
        u16, wxp16, xdbc);

    // selective scan -> gated y (f16)
    k_scan<<<dim3(DI / 256, BATCH), 256, 0, stream>>>(
        xz, u16, xdbc, dt_w, dt_b, A_log, Dpar, y16);

    // yo = y @ out_proj^T  (8192 x 256 x 1024): 32x64 per wave, 4 waves/block
    k_gemm_nt<2, 4, DM, DI><<<dim3(ROWS / 32, 1), 128, 0, stream>>>(
        y16, wout16, yo);

    // last-step logits + softmax -> d_out[:, COFF[s] : COFF[s]+outs]
    k_head<<<BATCH, 32, 0, stream>>>(yo, W2, b2, temp, dout, OUTSV[s], COFF[s]);
  }
}
